// AnomalyAttention_48412871360570
// MI455X (gfx1250) — matrix operand order repaired and checked by reading, not compiled, not run on hardware
//
#include <hip/hip_runtime.h>

// ---------------------------------------------------------------------------
// AnomalyAttention (banded attention, DIAG=64) for MI455X / gfx1250 (wave32)
// outputs: V [B,L,H,E] f32  ++  series [B,H,L,L] f32 (softmax matrix)
// Strategy: series is 268MB of mostly zeros -> NT zero-fill stream, then a
// band kernel using v_wmma_f32_16x16x32_f16 for QK^T and P*V.
// ---------------------------------------------------------------------------

typedef __attribute__((ext_vector_type(16))) _Float16 v16h;
typedef __attribute__((ext_vector_type(8)))  _Float16 v8h;
typedef __attribute__((ext_vector_type(8)))  float    v8f;
typedef __attribute__((ext_vector_type(4)))  float    v4f;

#define B_   2
#define L_   2048
#define H_   8
#define E_   64
#define DIAG_ 64
#define NT_  10            // 9 live j-tiles + 1 zero pad tile (K=160 for P*V)
#define BAND_COLS 160      // NT_*16
#define WPB  4             // waves per block

__device__ __forceinline__ v8f wmma_f16(v16h a, v16h b, v8f c) {
  // D = A(16x32 f16) * B(32x16 f16) + C(16x16 f32)
  return __builtin_amdgcn_wmma_f32_16x16x32_f16(
      false, a, false, b, (short)0, c, false, false);
}

__device__ __forceinline__ v8h cvt8(const float* __restrict__ p) {
  v4f a = *(const v4f*)p;
  v4f b = *(const v4f*)(p + 4);
  v8h r;
  r[0]=(_Float16)a[0]; r[1]=(_Float16)a[1]; r[2]=(_Float16)a[2]; r[3]=(_Float16)a[3];
  r[4]=(_Float16)b[0]; r[5]=(_Float16)b[1]; r[6]=(_Float16)b[2]; r[7]=(_Float16)b[3];
  return r;
}
__device__ __forceinline__ v16h cat16(v8h lo, v8h hi) {
  return __builtin_shufflevector(lo, hi, 0,1,2,3,4,5,6,7,8,9,10,11,12,13,14,15);
}
__device__ __forceinline__ v16h cvt16(const float* __restrict__ p) {
  return cat16(cvt8(p), cvt8(p + 8));
}

// --------------------------- zero-fill of `series` -------------------------
__global__ void series_zero_kernel(float* __restrict__ series, long n4) {
  long i = (long)blockIdx.x * blockDim.x + threadIdx.x;
  if (i < n4) {
    v4f z = {0.f, 0.f, 0.f, 0.f};
    __builtin_nontemporal_store(z, ((v4f*)series) + i);   // NT: write-once stream
  }
}

// ------------------------------ band kernel --------------------------------
__global__ __launch_bounds__(WPB * 32)
void anomaly_attn_kernel(const float* __restrict__ Q,
                         const float* __restrict__ K,
                         const float* __restrict__ V,
                         float* __restrict__ Vout,
                         float* __restrict__ series) {
  __shared__ __align__(16) _Float16 ldsP[WPB][16][BAND_COLS];  // 20KB / block

  const int tid  = threadIdx.x;
  const int wave = tid >> 5;
  const int lane = tid & 31;
  const int half = lane >> 4;   // which 16-lane half of the wave
  const int l16  = lane & 15;

  const int wid = blockIdx.x * WPB + wave;  // 0 .. B*H*(L/16)-1 = 2047
  const int i0  = (wid & 127) << 4;         // row block start
  const int h   = (wid >> 7) & 7;
  const int b   = wid >> 10;
  const int jbase = i0 - 64;                // first band column (tile aligned)

  // ---- Q operand in WMMA A-layout (16x32 f16), kept for all j-tiles ----
  // A layout: row M = l16 (both halves); K elems: half*16 + 0..15 (contiguous)
  const float* qrow = Q + (((size_t)(b * L_ + (i0 + l16))) * H_ + h) * E_;
  v16h aq0 = cvt16(qrow +  0 + half * 16);
  v16h aq1 = cvt16(qrow + 32 + half * 16);

  // ---- scores for the 9 live tiles (tile 9 = pad, fully masked) ----
  v8f S[NT_];
  #pragma unroll
  for (int t = 0; t < NT_; ++t) {
    const int j0 = jbase + t * 16;
    v8f s;
    if (t < 9 && j0 >= 0 && j0 < L_) {
      // B layout: col N = l16, K elems = half*16 + 0..15 (contiguous e of key row)
      const float* krow = K + (((size_t)(b * L_ + (j0 + l16))) * H_ + h) * E_;
      v16h bk0 = cvt16(krow + half * 16);
      v16h bk1 = cvt16(krow + 32 + half * 16);
      v8f c = {};
      c = wmma_f16(aq0, bk0, c);
      c = wmma_f16(aq1, bk1, c);
      #pragma unroll
      for (int r = 0; r < 8; ++r) {           // C layout: row = r + 8*half, col = l16
        int di = (i0 + r + half * 8) - (j0 + l16);
        s[r] = (di >= DIAG_ || di <= -DIAG_) ? -__builtin_inff()
                                             : c[r] * 0.125f;   // 1/sqrt(64)
      }
    } else {
      #pragma unroll
      for (int r = 0; r < 8; ++r) s[r] = -__builtin_inff();
    }
    S[t] = s;
  }

  // ---- softmax over the band (row lives on one 16-lane half-group) ----
  float inv[8], mrow[8];
  #pragma unroll
  for (int r = 0; r < 8; ++r) {
    float m = -__builtin_inff();
    #pragma unroll
    for (int t = 0; t < NT_; ++t) m = fmaxf(m, S[t][r]);
    #pragma unroll
    for (int x = 1; x < 16; x <<= 1) m = fmaxf(m, __shfl_xor(m, x, 32));
    mrow[r] = m;
  }
  #pragma unroll
  for (int r = 0; r < 8; ++r) {
    float sum = 0.f;
    #pragma unroll
    for (int t = 0; t < NT_; ++t) { float p = __expf(S[t][r] - mrow[r]); S[t][r] = p; sum += p; }
    #pragma unroll
    for (int x = 1; x < 16; x <<= 1) sum += __shfl_xor(sum, x, 32);
    inv[r] = __builtin_amdgcn_rcpf(sum);
  }

  // ---- emit P: f16 copy to LDS (for P*V) + NT f32 stores into series ----
  const long srow0 = (long)(b * H_ + h) * L_;
  #pragma unroll
  for (int t = 0; t < NT_; ++t) {
    const int j0 = jbase + t * 16;
    const bool wr = (t < 9) && (j0 >= 0) && (j0 < L_);  // pad/out tiles are exact 0s
    #pragma unroll
    for (int r = 0; r < 8; ++r) {
      const int irow = r + half * 8;
      float p = S[t][r] * inv[r];
      ldsP[wave][irow][t * 16 + l16] = (_Float16)p;
      if (wr) {
        long off = (srow0 + (i0 + irow)) * (long)L_ + (j0 + l16);
        __builtin_nontemporal_store(p, series + off);   // lanes 0-15 -> 64B runs
      }
    }
  }
  // same-wave LDS RAW: compiler inserts s_wait_dscnt; no cross-wave sharing.

  // ---- P(16x160) * V(160x64): 5 K-chunks x 4 d-chunks of 16x16x32 WMMA ----
  v8f acc[4];
  #pragma unroll
  for (int d = 0; d < 4; ++d) acc[d] = (v8f){};
  #pragma unroll
  for (int c = 0; c < 5; ++c) {
    // A operand from LDS (row = l16, K = half*16 + 0..15 contiguous)
    v16h pa = *(const v16h*)&ldsP[wave][l16][c * 32 + half * 16];
    #pragma unroll
    for (int d = 0; d < 4; ++d) {
      // B operand: col d = d*16 + l16 fixed, 16 consecutive j rows per lane;
      // lanes 0-15 share j and cover consecutive d -> 64B-coalesced loads.
      const int dc = d * 16 + l16;
      v16h bv;
      #pragma unroll
      for (int k = 0; k < 16; ++k) {
        int j = jbase + c * 32 + half * 16 + k;
        j = j < 0 ? 0 : (j > L_ - 1 ? L_ - 1 : j);      // clamped rows hit P==0
        bv[k] = (_Float16)V[(((size_t)(b * L_ + j)) * H_ + h) * E_ + dc];
      }
      acc[d] = wmma_f16(pa, bv, acc[d]);
    }
  }

  // ---- store V output [B,L,H,E] (coalesced 64B per VGPR row) ----
  #pragma unroll
  for (int d = 0; d < 4; ++d) {
    #pragma unroll
    for (int r = 0; r < 8; ++r) {
      const int i = i0 + r + half * 8;
      size_t off = (((size_t)(b * L_ + i)) * H_ + h) * E_ + d * 16 + l16;
      __builtin_nontemporal_store(acc[d][r], Vout + off);
    }
  }
}

// ---------------------------------------------------------------------------
extern "C" void kernel_launch(void* const* d_in, const int* in_sizes, int n_in,
                              void* d_out, int out_size, void* d_ws, size_t ws_size,
                              hipStream_t stream) {
  const float* Q = (const float*)d_in[0];
  const float* K = (const float*)d_in[1];
  const float* V = (const float*)d_in[2];
  // d_in[3] = sigma, d_in[4] = attn_mask: unused by the reference computation.

  float* Vout   = (float*)d_out;
  float* series = Vout + (size_t)B_ * L_ * H_ * E_;   // 2,097,152 floats in

  // 1) stream zeros over series (268 MB, NT float4)
  const long n4 = ((long)B_ * H_ * L_ * L_) / 4;      // 16,777,216 float4
  series_zero_kernel<<<(unsigned)((n4 + 255) / 256), 256, 0, stream>>>(series, n4);

  // 2) band attention: one wave per (b, h, 16-row block)
  const int nWaves = B_ * H_ * (L_ / 16);             // 2048
  anomaly_attn_kernel<<<nWaves / WPB, WPB * 32, 0, stream>>>(Q, K, V, Vout, series);
}
